// GCNLayer_13168369730080
// MI455X (gfx1250) — matrix operand order repaired and checked by reading, not compiled, not run on hardware
//
#include <hip/hip_runtime.h>

typedef __attribute__((ext_vector_type(16))) __bf16 v16bf;
typedef __attribute__((ext_vector_type(8)))  __bf16 v8bf;
typedef __attribute__((ext_vector_type(8)))  float  v8f;
typedef __attribute__((ext_vector_type(4)))  unsigned int v4u;
typedef __attribute__((ext_vector_type(8)))  int v8i;
typedef __attribute__((ext_vector_type(4)))  int v4i;

union V16U { v16bf v; v8bf h[2]; };

constexpr int Dm = 256;   // feature dim == out_dim
constexpr int Nn = 256;   // nodes
// B = 4 ; rows = B*N = 1024

static __device__ __forceinline__ unsigned short f32_bf16(float f) {
    unsigned int u = __float_as_uint(f);
    u += 0x7FFFu + ((u >> 16) & 1u);   // round-to-nearest-even
    return (unsigned short)(u >> 16);
}

// ---------------------------------------------------------------------------
// K0: weight transpose+convert to bf16 (wT[n*D+d] = w[d*D+n]) and x -> bf16
// blocks 0..767: three 256x256 weight mats (one output row per block)
// blocks 768..1791: x convert (1024 rows)
// ---------------------------------------------------------------------------
__global__ void convert_kernel(const float* __restrict__ x,
                               const float* __restrict__ wA,
                               const float* __restrict__ wVi,
                               const float* __restrict__ wVj,
                               unsigned short* __restrict__ wTA,
                               unsigned short* __restrict__ wTVi,
                               unsigned short* __restrict__ wTVj,
                               unsigned short* __restrict__ xb) {
    int bid = blockIdx.x, t = threadIdx.x;
    if (bid < 768) {
        int mat = bid >> 8, n = bid & 255;
        const float* src = (mat == 0) ? wA : ((mat == 1) ? wVi : wVj);
        unsigned short* dst = (mat == 0) ? wTA : ((mat == 1) ? wTVi : wTVj);
        dst[n * Dm + t] = f32_bf16(src[t * Dm + n]);
    } else {
        int idx = (bid - 768) * Dm + t;
        xb[idx] = f32_bf16(x[idx]);
    }
}

// ---------------------------------------------------------------------------
// K1: xi_b = x @ w_vi + bias_h ; xj = x @ w_vj   via WMMA bf16
// 2048 wave-tasks: which(2) x mtile(64) x ntile(16); 256 blocks x 8 waves
// ---------------------------------------------------------------------------
__global__ void proj_kernel(const unsigned short* __restrict__ xb,
                            const unsigned short* __restrict__ wTVi,
                            const unsigned short* __restrict__ wTVj,
                            const float* __restrict__ bias,
                            float* __restrict__ xi_b,
                            float* __restrict__ xjw) {
    int wid  = blockIdx.x * 8 + (threadIdx.x >> 5);
    int lane = threadIdx.x & 31;
    int which = wid >> 10;           // 0 = vi, 1 = vj
    int rem   = wid & 1023;
    int m0 = (rem >> 4) << 4;
    int n0 = (rem & 15) << 4;
    int lr = lane & 15, lh = lane >> 4;

    const unsigned short* wT = which ? wTVj : wTVi;
    const unsigned short* arow = xb + (m0 + lr) * Dm;
    const unsigned short* brow = wT + (n0 + lr) * Dm;

    v8f acc = {0.f, 0.f, 0.f, 0.f, 0.f, 0.f, 0.f, 0.f};
    for (int k0 = 0; k0 < Dm; k0 += 32) {
        // A 16x32 bf16 layout: lane holds K = {lh*8..+7} and {16+lh*8..+7}
        V16U a;
        a.h[0] = *(const v8bf*)(arow + k0 + lh * 8);
        a.h[1] = *(const v8bf*)(arow + k0 + 16 + lh * 8);
        // B 32x16 bf16 layout: lane holds K = {lh*8..+7} and {16+lh*8..+7}
        V16U bu;
        bu.h[0] = *(const v8bf*)(brow + k0 + lh * 8);
        bu.h[1] = *(const v8bf*)(brow + k0 + 16 + lh * 8);
        v16bf b = bu.v;
        acc = __builtin_amdgcn_wmma_f32_16x16x32_bf16(
            false, a.v, false, b, (short)0, acc, false, false);
    }
    float* outp = which ? xjw : xi_b;
    float bv = which ? 0.f : bias[n0 + lr];
    for (int v = 0; v < 8; ++v)
        outp[(m0 + v + lh * 8) * Dm + (n0 + lr)] = acc[v] + bv;
}

// ---------------------------------------------------------------------------
// K2: fused per-(b,i) kernel. 1024 blocks x 256 threads (8 waves).
// LDS: wT_alpha bf16 128KB | H tiles 8x(16x256) bf16 64KB | xi/adj/AH 3KB
// wT preload done by the Tensor Data Mover (one D# descriptor, wave 0).
// ---------------------------------------------------------------------------
__global__ void fused_kernel(const float* __restrict__ alpha,
                             const float* __restrict__ adjm,
                             const float* __restrict__ xi_b,
                             const float* __restrict__ xjw,
                             const unsigned short* __restrict__ wTA,
                             const float* __restrict__ w_node,
                             float* __restrict__ new_x,
                             float* __restrict__ new_alpha) {
    extern __shared__ char smem[];
    unsigned short* sh_wT = (unsigned short*)smem;             // 131072 B
    unsigned short* sh_H  = (unsigned short*)(smem + 131072);  //  65536 B
    float* sh_xi  = (float*)(smem + 196608);
    float* sh_adj = (float*)(smem + 197632);
    float* sh_AH  = (float*)(smem + 198656);

    int tid = threadIdx.x;
    int wave = tid >> 5, lane = tid & 31;
    int bi = blockIdx.x;                      // b*256 + i
    int bj0base = (bi >> 8) << 8;             // b*256

#if __has_builtin(__builtin_amdgcn_tensor_load_to_lds)
    // ---- TDM: one DMA descriptor moves 131072 B (16384 x 8B) to LDS ----
    if (wave == 0) {
        unsigned long long ga = (unsigned long long)wTA;
        unsigned ldsa = (unsigned)(size_t)(void*)sh_wT;
        v4u g0;
        g0[0] = 1u;                                   // count=1, user descriptor
        g0[1] = ldsa;                                 // lds_addr
        g0[2] = (unsigned)ga;                         // global_addr[31:0] @bit64
        g0[3] = (unsigned)((ga >> 32) & 0x1FFFFFFu)   // global_addr[56:32]
                | 0x80000000u;                        // type=2 ("image")
        v8i g1;
        g1[0] = 0x30000;                              // data_size=3 (8B), mask=0
        g1[1] = (int)(16384u << 16);                  // tensor_dim0[15:0]=16384
        g1[2] = (int)(1u << 16);                      // dim0 hi=0, tensor_dim1=1
        g1[3] = (int)(16384u << 16);                  // dim1 hi=0, tile_dim0=16384
        g1[4] = 0;                                    // tile_dim1=0, tile_dim2=0
        g1[5] = 16384;                                // tensor_dim0_stride lo
        g1[6] = 0;                                    // stride hi, dim1_stride lo
        g1[7] = 0;
        v4i gz4 = {0, 0, 0, 0};                       // groups 2/3 unused (<=2D)
        v8i gz8 = {0, 0, 0, 0, 0, 0, 0, 0};
        __builtin_amdgcn_tensor_load_to_lds(g0, g1, gz4, gz4, gz8, 0);
        __builtin_amdgcn_s_wait_tensorcnt(0);
    }
#else
    {   // fallback: plain preload, 8192 x uint4
        const uint4* src = (const uint4*)wTA;
        uint4* dst = (uint4*)sh_wT;
        for (int k = tid; k < 8192; k += 256) dst[k] = src[k];
    }
#endif
    sh_xi[tid]  = xi_b[bi * Dm + tid];
    sh_adj[tid] = adjm[bi * Nn + tid];
    sh_AH[tid]  = 0.f;
    __syncthreads();

    int c0 = lane * 8;
    float xir[8];
    for (int c = 0; c < 8; ++c) xir[c] = sh_xi[c0 + c];

    int lr = lane & 15, lh = lane >> 4;

    for (int jt = wave; jt < 16; jt += 8) {
        int j0 = jt * 16;
        float ah[8];
        for (int c = 0; c < 8; ++c) ah[c] = 0.f;

        // ---- build relu'd bf16 H tile (16 rows x 256) in this wave's LDS ----
        for (int r = 0; r < 16; ++r) {
            const float* ap = alpha + (((long)bi * Nn) + (j0 + r)) * (long)Dm + c0;
            __builtin_prefetch(ap + Dm, 0, 0);
            float4 a0 = ((const float4*)ap)[0];
            float4 a1 = ((const float4*)ap)[1];
            const float* xp = xjw + (bj0base + j0 + r) * Dm + c0;
            float4 x0 = ((const float4*)xp)[0];
            float4 x1 = ((const float4*)xp)[1];
            float adjj = sh_adj[j0 + r];

            float h[8];
            h[0] = fmaxf(xir[0] + x0.x + a0.x, 0.f);
            h[1] = fmaxf(xir[1] + x0.y + a0.y, 0.f);
            h[2] = fmaxf(xir[2] + x0.z + a0.z, 0.f);
            h[3] = fmaxf(xir[3] + x0.w + a0.w, 0.f);
            h[4] = fmaxf(xir[4] + x1.x + a1.x, 0.f);
            h[5] = fmaxf(xir[5] + x1.y + a1.y, 0.f);
            h[6] = fmaxf(xir[6] + x1.z + a1.z, 0.f);
            h[7] = fmaxf(xir[7] + x1.w + a1.w, 0.f);
            for (int c = 0; c < 8; ++c) ah[c] = fmaf(adjj, h[c], ah[c]);

            uint4 p;
            p.x = (unsigned)f32_bf16(h[0]) | ((unsigned)f32_bf16(h[1]) << 16);
            p.y = (unsigned)f32_bf16(h[2]) | ((unsigned)f32_bf16(h[3]) << 16);
            p.z = (unsigned)f32_bf16(h[4]) | ((unsigned)f32_bf16(h[5]) << 16);
            p.w = (unsigned)f32_bf16(h[6]) | ((unsigned)f32_bf16(h[7]) << 16);
            *(uint4*)(sh_H + wave * 4096 + r * 256 + c0) = p;      // ds_store_b128
        }
        for (int c = 0; c < 8; ++c) atomicAdd(&sh_AH[c0 + c], ah[c]); // ds_add_f32

        // ---- hoist the 8 A-fragments (whole K=256) into VGPRs once ----
        const unsigned short* Arow = sh_H + wave * 4096 + lr * 256;
        V16U afr[8];
#pragma unroll
        for (int kk = 0; kk < 8; ++kk) {
            int k0 = kk * 32;
            afr[kk].h[0] = *(const v8bf*)(Arow + k0 + lh * 8);
            afr[kk].h[1] = *(const v8bf*)(Arow + k0 + 16 + lh * 8);
        }

        // ---- new_alpha tile = relu(H_tile @ w_alpha) via WMMA bf16 ----
        long outbase = ((long)bi * Nn + j0) * Dm;
        for (int nt = 0; nt < 16; ++nt) {
            const unsigned short* Brow = sh_wT + (nt * 16 + lr) * Dm;
            v8f acc = {0.f, 0.f, 0.f, 0.f, 0.f, 0.f, 0.f, 0.f};
#pragma unroll
            for (int kk = 0; kk < 8; ++kk) {
                V16U bu;
                bu.h[0] = *(const v8bf*)(Brow + kk * 32 + lh * 8);
                bu.h[1] = *(const v8bf*)(Brow + kk * 32 + 16 + lh * 8);
                v16bf b = bu.v;
                acc = __builtin_amdgcn_wmma_f32_16x16x32_bf16(
                    false, afr[kk].v, false, b, (short)0, acc, false, false);
            }
            int n = nt * 16 + lr;
            for (int v = 0; v < 8; ++v)
                new_alpha[outbase + (long)(v + lh * 8) * Dm + n] = fmaxf(acc[v], 0.f);
        }
    }
    __syncthreads();

    // ---- new_x[b,i,:] = relu(AH @ w_node) ; thread t -> column t ----
    float s = 0.f;
    for (int d = 0; d < Dm; ++d)
        s = fmaf(sh_AH[d], w_node[d * Dm + tid], s);
    new_x[bi * Dm + tid] = fmaxf(s, 0.f);
}

// ---------------------------------------------------------------------------
extern "C" void kernel_launch(void* const* d_in, const int* in_sizes, int n_in,
                              void* d_out, int out_size, void* d_ws, size_t ws_size,
                              hipStream_t stream) {
    const float* x       = (const float*)d_in[0];
    const float* alpha   = (const float*)d_in[1];
    const float* adj     = (const float*)d_in[2];
    // d_in[3] box_num: unused by the reference computation
    const float* w_alpha = (const float*)d_in[4];
    const float* w_vi    = (const float*)d_in[5];
    const float* w_vj    = (const float*)d_in[6];
    const float* bias_h  = (const float*)d_in[7];
    const float* w_node  = (const float*)d_in[8];

    float* new_x     = (float*)d_out;
    float* new_alpha = new_x + 4 * 256 * 256;   // outputs concatenated flat

    char* ws = (char*)d_ws;
    unsigned short* wTA  = (unsigned short*)(ws);             // 128 KB
    unsigned short* wTVi = (unsigned short*)(ws + 131072);    // 128 KB
    unsigned short* wTVj = (unsigned short*)(ws + 262144);    // 128 KB
    unsigned short* xb   = (unsigned short*)(ws + 393216);    // 512 KB
    float* xi_b          = (float*)(ws + 917504);             //   1 MB
    float* xjw           = (float*)(ws + 1966080);            //   1 MB

    convert_kernel<<<1792, 256, 0, stream>>>(x, w_alpha, w_vi, w_vj,
                                             wTA, wTVi, wTVj, xb);
    proj_kernel<<<256, 256, 0, stream>>>(xb, wTVi, wTVj, bias_h, xi_b, xjw);
    fused_kernel<<<1024, 256, 199680, stream>>>(alpha, adj, xi_b, xjw,
                                                wTA, w_node, new_x, new_alpha);
}
